// MultiHeadedAttentionWithCache_61847529062416
// MI455X (gfx1250) — matrix operand order repaired and checked by reading, not compiled, not run on hardware
//
#include <hip/hip_runtime.h>
#include <hip/hip_bf16.h>

typedef __attribute__((ext_vector_type(16))) _Float16 v16h;
typedef __attribute__((ext_vector_type(8)))  float    v8f;
typedef __attribute__((ext_vector_type(4)))  float    f32x4;   // native vec for NT builtins

// Problem constants
constexpr int  Dm  = 1024;
constexpr int  Bb  = 16, Qq = 16, Hh = 16, DHd = 64, Cc = 4096;
constexpr int  Lk  = Cc + Qq;                 // 4112 keys
constexpr long OUT_O = 0;
constexpr long OUT_K = (long)Bb * Qq * Dm;    // 262144
constexpr long KVSZ  = (long)Bb * Hh * Lk * DHd; // 67371008
constexpr long OUT_V = OUT_K + KVSZ;

__device__ __forceinline__ v8f wmma_f16(v16h a, v16h b, v8f c) {
  // v_wmma_f32_16x16x32_f16, fp32 accumulate
  return __builtin_amdgcn_wmma_f32_16x16x32_f16(
      /*neg_a=*/false, a, /*neg_b=*/false, b,
      /*c_mod=*/(short)0, c, /*reuse_a=*/false, /*reuse_b=*/false);
}

// ---------------------------------------------------------------------------
// Kernel 0: stream cache_k/cache_v into k_up/v_up regions of d_out (B128 + NT)
// ---------------------------------------------------------------------------
__global__ __launch_bounds__(256) void mha_copy_cache(
    const f32x4* __restrict__ ck, const f32x4* __restrict__ cv,
    f32x4* __restrict__ outp) {
  long idx = (long)blockIdx.x * 256 + threadIdx.x;   // 0 .. 33554431
  int tensor = (int)(idx >> 24);                     // 16777216 f4 per tensor
  long rem   = idx & 0xFFFFFF;
  long bh    = rem >> 16;                            // C*DH/4 = 65536 per (b,h)
  long w     = rem & 0xFFFF;
  const f32x4* src = tensor ? cv : ck;
  f32x4* dst = outp + (tensor ? (OUT_V >> 2) : (OUT_K >> 2));
  f32x4 v = __builtin_nontemporal_load(src + bh * 65536 + w);
  __builtin_nontemporal_store(v, dst + bh * 65792 + w);  // (C+Q)*DH/4 = 65792
}

// ---------------------------------------------------------------------------
// Kernel 1: Q/K/V projections, one wave per 16x16 tile, WMMA f16 over K=1024.
//   Q -> ws_q in [B,H,Q,DH];  K/V -> tail rows (C..C+15) of k_up / v_up.
// ---------------------------------------------------------------------------
__global__ __launch_bounds__(32) void mha_proj_qkv(
    const float* __restrict__ q, const float* __restrict__ k,
    const float* __restrict__ v,
    const float* __restrict__ Wq, const float* __restrict__ bq,
    const float* __restrict__ Wk, const float* __restrict__ bk,
    const float* __restrict__ Wv, const float* __restrict__ bv,
    float* __restrict__ wsq, float* __restrict__ outp) {
  int t     = blockIdx.x;          // 0..3071
  int which = t >> 10;             // 0=Q 1=K 2=V
  int tile  = t & 1023;
  int m0 = (tile >> 6) * 16, n0 = (tile & 63) * 16;
  const float* X    = which == 0 ? q  : (which == 1 ? k  : v);
  const float* W    = which == 0 ? Wq : (which == 1 ? Wk : Wv);
  const float* bias = which == 0 ? bq : (which == 1 ? bk : bv);

  int lane = threadIdx.x, lr = lane & 15, half = lane >> 4;
  const float* xr = X + (long)(m0 + lr) * Dm;   // A row for this lane
  const float* wc = W + (n0 + lr);              // B column for this lane
  v8f acc = {};
  for (int k0 = 0; k0 < Dm; k0 += 32) {
    v16h a, b;
    int ka = k0 + half * 8;       // A layout: K interleaved 8/16/24 per half
    int kb = k0 + half * 8;       // B layout: K interleaved 8/16/24 per half
#pragma unroll
    for (int i = 0; i < 8; ++i) {
      a[i]     = (_Float16)xr[ka + i];
      a[8 + i] = (_Float16)xr[ka + 16 + i];
    }
#pragma unroll
    for (int i = 0; i < 16; ++i)
      b[i] = (_Float16)wc[(long)(kb + (i & 7) + ((i >> 3) << 4)) * Dm];
    acc = wmma_f16(a, b, acc);
  }
  int col = n0 + lr;
  float bconst = bias[col];
  int h = col >> 6, dh = col & 63;
#pragma unroll
  for (int r = 0; r < 8; ++r) {
    int row = m0 + half * 8 + r;        // 0..255 = b*16 + qrow
    int b_ = row >> 4, qrow = row & 15;
    float val = acc[r] + bconst;
    if (which == 0) {
      wsq[(((long)(b_ * Hh + h) * Qq) + qrow) * DHd + dh] = val;
    } else {
      float* base = outp + (which == 2 ? OUT_V : OUT_K);
      base[((long)(b_ * Hh + h) * Lk + (Cc + qrow)) * DHd + dh] = val;
    }
  }
}

// ---------------------------------------------------------------------------
// Kernel 2: flash attention, one wave per (b,h). Online softmax, fp32 accum.
// ---------------------------------------------------------------------------
__global__ __launch_bounds__(32) void mha_attn(
    const float* __restrict__ wsq, const int* __restrict__ mask,
    const float* __restrict__ outp, float* __restrict__ wsx) {
  __shared__ float ldsP[16][33];       // P transpose staging (padded)
  int bh = blockIdx.x;                 // 0..255
  int b_ = bh >> 4, h = bh & 15;
  int lane = threadIdx.x, lr = lane & 15, half = lane >> 4;

  const float* kb_base = outp + OUT_K + (long)bh * Lk * DHd;
  const float* vb_base = outp + OUT_V + (long)bh * Lk * DHd;
  const float* qrow    = wsq + ((long)bh * Qq + lr) * DHd;
  const int*   mrow    = mask + (long)b_ * Qq * Lk;

  // Q A-fragments for DH chunks [0,32) and [32,64)
  v16h a0, a1;
#pragma unroll
  for (int i = 0; i < 8; ++i) {
    a0[i]     = (_Float16)qrow[half * 8 + i];
    a0[8 + i] = (_Float16)qrow[16 + half * 8 + i];
    a1[i]     = (_Float16)qrow[32 + half * 8 + i];
    a1[8 + i] = (_Float16)qrow[48 + half * 8 + i];
  }

  v8f O0 = {}, O1 = {}, O2 = {}, O3 = {};
  float m[8], l[8];
#pragma unroll
  for (int r = 0; r < 8; ++r) { m[r] = -3.0e30f; l[r] = 0.0f; }
  const float scale = 0.125f;  // 1/sqrt(64)

  for (int j = 0; j < 129; ++j) {      // 129*32 = 4128 >= 4112 keys
    int n0 = j * 32;
    int key0 = n0 + lr, key1 = n0 + 16 + lr;
    int key0c = key0 < Lk ? key0 : Lk - 1;
    int key1c = key1 < Lk ? key1 : Lk - 1;
    const float* kr0 = kb_base + (long)key0c * DHd;
    const float* kr1 = kb_base + (long)key1c * DHd;

    // B fragments: column = key row (contiguous DH), K interleaved 8/16/24 per half
    v16h b00, b01, b10, b11;
#pragma unroll
    for (int i = 0; i < 16; ++i) {
      b00[i] = (_Float16)kr0[half * 8 + (i & 7) + ((i >> 3) << 4)];
      b01[i] = (_Float16)kr0[32 + half * 8 + (i & 7) + ((i >> 3) << 4)];
      b10[i] = (_Float16)kr1[half * 8 + (i & 7) + ((i >> 3) << 4)];
      b11[i] = (_Float16)kr1[32 + half * 8 + (i & 7) + ((i >> 3) << 4)];
    }
    v8f S0 = {}, S1 = {};
    S0 = wmma_f16(a0, b00, S0); S0 = wmma_f16(a1, b01, S0);
    S1 = wmma_f16(a0, b10, S1); S1 = wmma_f16(a1, b11, S1);

    // scale + mask (+ tail padding) in C-layout
    float s0[8], s1[8], p0[8], p1[8];
#pragma unroll
    for (int r = 0; r < 8; ++r) {
      int M = half * 8 + r;
      bool ok0 = (key0 < Lk) && (mrow[(long)M * Lk + key0] != 0);
      bool ok1 = (key1 < Lk) && (mrow[(long)M * Lk + key1] != 0);
      s0[r] = ok0 ? S0[r] * scale : -3.0e30f;
      s1[r] = ok1 ? S1[r] * scale : -3.0e30f;
    }
    // online softmax: row reductions across the 16 lanes of each half
#pragma unroll
    for (int r = 0; r < 8; ++r) {
      float vmax = fmaxf(s0[r], s1[r]);
      vmax = fmaxf(vmax, __shfl_xor(vmax, 1, 16));
      vmax = fmaxf(vmax, __shfl_xor(vmax, 2, 16));
      vmax = fmaxf(vmax, __shfl_xor(vmax, 4, 16));
      vmax = fmaxf(vmax, __shfl_xor(vmax, 8, 16));
      float mn = fmaxf(m[r], vmax);
      float alpha = __expf(m[r] - mn);
      p0[r] = __expf(s0[r] - mn);
      p1[r] = __expf(s1[r] - mn);
      float bs = p0[r] + p1[r];
      bs += __shfl_xor(bs, 1, 16);
      bs += __shfl_xor(bs, 2, 16);
      bs += __shfl_xor(bs, 4, 16);
      bs += __shfl_xor(bs, 8, 16);
      l[r] = l[r] * alpha + bs;
      m[r] = mn;
      O0[r] *= alpha; O1[r] *= alpha; O2[r] *= alpha; O3[r] *= alpha;
    }

    // P: C-layout -> LDS -> A-layout fragment
    __syncthreads();
#pragma unroll
    for (int r = 0; r < 8; ++r) {
      ldsP[half * 8 + r][lr]      = p0[r];
      ldsP[half * 8 + r][16 + lr] = p1[r];
    }
    __syncthreads();
    v16h pa;
#pragma unroll
    for (int i = 0; i < 8; ++i) {
      pa[i]     = (_Float16)ldsP[lr][half * 8 + i];
      pa[8 + i] = (_Float16)ldsP[lr][16 + half * 8 + i];
    }

    // O += P @ V   (4 N-tiles of 16 over DH=64)
#pragma unroll
    for (int t = 0; t < 4; ++t) {
      int n = t * 16 + lr;  // dh column
      v16h vb;
#pragma unroll
      for (int i = 0; i < 16; ++i) {
        int kk = n0 + half * 8 + (i & 7) + ((i >> 3) << 4);
        int kkc = kk < Lk ? kk : Lk - 1;  // clamped; P==0 for padding keys
        vb[i] = (_Float16)vb_base[(long)kkc * DHd + n];
      }
      if      (t == 0) O0 = wmma_f16(pa, vb, O0);
      else if (t == 1) O1 = wmma_f16(pa, vb, O1);
      else if (t == 2) O2 = wmma_f16(pa, vb, O2);
      else             O3 = wmma_f16(pa, vb, O3);
    }
  }

  // finalize: x[b, q, h*64 + dh]  (rows of [256 x 1024])
#pragma unroll
  for (int r = 0; r < 8; ++r) {
    float inv = 1.0f / l[r];
    int M = half * 8 + r;  // query row
    long rowbase = (long)(b_ * Qq + M) * Dm + h * DHd;
    wsx[rowbase +  0 + lr] = O0[r] * inv;
    wsx[rowbase + 16 + lr] = O1[r] * inv;
    wsx[rowbase + 32 + lr] = O2[r] * inv;
    wsx[rowbase + 48 + lr] = O3[r] * inv;
  }
}

// ---------------------------------------------------------------------------
// Kernel 3: out = x @ Wo + bo
// ---------------------------------------------------------------------------
__global__ __launch_bounds__(32) void mha_oproj(
    const float* __restrict__ wsx, const float* __restrict__ Wo,
    const float* __restrict__ bo, float* __restrict__ outp) {
  int tile = blockIdx.x;               // 0..1023
  int m0 = (tile >> 6) * 16, n0 = (tile & 63) * 16;
  int lane = threadIdx.x, lr = lane & 15, half = lane >> 4;
  const float* xr = wsx + (long)(m0 + lr) * Dm;
  const float* wc = Wo + (n0 + lr);
  v8f acc = {};
  for (int k0 = 0; k0 < Dm; k0 += 32) {
    v16h a, b;
    int ka = k0 + half * 8;
    int kb = k0 + half * 8;
#pragma unroll
    for (int i = 0; i < 8; ++i) {
      a[i]     = (_Float16)xr[ka + i];
      a[8 + i] = (_Float16)xr[ka + 16 + i];
    }
#pragma unroll
    for (int i = 0; i < 16; ++i)
      b[i] = (_Float16)wc[(long)(kb + (i & 7) + ((i >> 3) << 4)) * Dm];
    acc = wmma_f16(a, b, acc);
  }
  int col = n0 + lr;
  float bconst = bo[col];
#pragma unroll
  for (int r = 0; r < 8; ++r)
    outp[(long)(m0 + half * 8 + r) * Dm + col] = acc[r] + bconst;
}

// ---------------------------------------------------------------------------
extern "C" void kernel_launch(void* const* d_in, const int* in_sizes, int n_in,
                              void* d_out, int out_size, void* d_ws, size_t ws_size,
                              hipStream_t stream) {
  const float* query   = (const float*)d_in[0];
  const float* key     = (const float*)d_in[1];
  const float* value   = (const float*)d_in[2];
  const int*   mask    = (const int*)  d_in[3];
  const float* cache_k = (const float*)d_in[4];
  const float* cache_v = (const float*)d_in[5];
  const float* Wq = (const float*)d_in[6];
  const float* bq = (const float*)d_in[7];
  const float* Wk = (const float*)d_in[8];
  const float* bk = (const float*)d_in[9];
  const float* Wv = (const float*)d_in[10];
  const float* bv = (const float*)d_in[11];
  const float* Wo = (const float*)d_in[12];
  const float* bo = (const float*)d_in[13];

  float* outp = (float*)d_out;
  float* wsq  = (float*)d_ws;            // 256*1024 floats (1 MB)
  float* wsx  = wsq + (long)Bb * Qq * Dm; // 256*1024 floats (1 MB)

  // 1) stream KV cache into k_up/v_up (bandwidth-bound path)
  mha_copy_cache<<<131072, 256, 0, stream>>>(
      (const f32x4*)cache_k, (const f32x4*)cache_v, (f32x4*)outp);
  // 2) QKV projections (WMMA)
  mha_proj_qkv<<<3072, 32, 0, stream>>>(query, key, value,
                                        Wq, bq, Wk, bk, Wv, bv, wsq, outp);
  // 3) flash attention per (b,h) (WMMA + online softmax)
  mha_attn<<<256, 32, 0, stream>>>(wsq, mask, outp, wsx);
  // 4) output projection (WMMA)
  mha_oproj<<<1024, 32, 0, stream>>>(wsx, Wo, bo, outp);
}
